// Experts_36146444763952
// MI455X (gfx1250) — matrix operand order repaired and checked by reading, not compiled, not run on hardware
//
#include <hip/hip_runtime.h>

// MoE grouped FFN: y = gelu(x_e @ W1_e + b1_e) @ W2_e + b2_e, per expert chunk.
// E=8, B=4, C=1024, D=1024, F=4096.  M per expert = B*C = 4096.

#define NE     8
#define CAP    1024
#define Dm     1024
#define Ff     4096
#define MPE    4096        // rows per expert (B*C)
#define FCHUNK 256         // F-tile per fused iteration

typedef __attribute__((ext_vector_type(16))) __bf16       v16bf;
typedef __attribute__((ext_vector_type(2)))  __bf16       v2bf;
typedef __attribute__((ext_vector_type(8)))  float        v8f;
typedef __attribute__((ext_vector_type(2)))  float        v2f;
typedef __attribute__((ext_vector_type(4)))  unsigned int v4u;
typedef __attribute__((ext_vector_type(8)))  unsigned int v8u;

// LDS layout
#define SX_STRIDE   516                        // uints per X row: 512 bf16-pairs + 4 pad
#define SH_STRIDE_H 264                        // halves per H row: 256 + 8 pad (528B = 33*16)
#define SH_STRIDE_U 132
#define SW1_STRIDE  36                         // uints per W1t row: 32 kpairs + 4 pad
#define SW2_STRIDE  20                         // uints per W2t row: 16 kpairs + 4 pad
#define SX_BYTES    (16 * SX_STRIDE * 4)       // 33024
#define SH_BYTES    (16 * SH_STRIDE_H * 2)     // 8448
#define SW1_BYTES   (FCHUNK * SW1_STRIDE * 4)  // 36864 (x2 ping-pong = 73728)
#define SW_BYTES    (1024 * SW2_STRIDE * 4)    // 81920 >= 2*SW1_BYTES, union region
#define SMEM_BYTES  (SX_BYTES + SH_BYTES + SW_BYTES)  // 123392 B -> 2 WGs / 320KB WGP

// Packed fp32->bf16 (RNE); backend lowers to packed v_cvt on gfx1250.
__device__ __forceinline__ unsigned int pack2bf(float a, float b) {
    v2f  t = {a, b};
    v2bf r = __builtin_convertvector(t, v2bf);
    return __builtin_bit_cast(unsigned int, r);
}

__device__ __forceinline__ float gelu_tanh(float x) {
    float u = 0.7978845608028654f * (x + 0.044715f * x * x * x);
    float e = __expf(2.0f * u);
    float t = (e - 1.0f) / (e + 1.0f);
    return 0.5f * x * (1.0f + t);
}

// A fragment (16x32 bf16): lane<16 holds K {kb..kb+7, kb+16..kb+23}, lane>=16 holds +8.
__device__ __forceinline__ v16bf load_frag_a(const unsigned int* sA, int rowStrideU,
                                             int row, int kpairBase, int lane) {
    const unsigned int* p = sA + row * rowStrideU + kpairBase + ((lane >> 4) & 1) * 4;
    v4u lo = *(const v4u*)(p);
    v4u hi = *(const v4u*)(p + 8);
    v8u c  = __builtin_shufflevector(lo, hi, 0, 1, 2, 3, 4, 5, 6, 7);
    return __builtin_bit_cast(v16bf, c);
}

// B fragment (32x16 bf16): lane -> N row; lane<16 K {0..7, 16..23}, lane>=16 +8.
// Weights stored K-transposed: sW[n][kpair] -> two contiguous uint4 loads.
__device__ __forceinline__ v16bf load_frag_b(const unsigned int* sW, int rowStrideU,
                                             int nrow, int kpairBase, int lane) {
    const unsigned int* p = sW + nrow * rowStrideU + kpairBase + ((lane >> 4) & 1) * 4;
    v4u lo = *(const v4u*)(p);
    v4u hi = *(const v4u*)(p + 8);
    v8u c  = __builtin_shufflevector(lo, hi, 0, 1, 2, 3, 4, 5, 6, 7);
    return __builtin_bit_cast(v16bf, c);
}

// Stage W1 [64k x FCHUNK f] transposed -> buf[f][kpair]  (coalesced over f)
__device__ __forceinline__ void stage_w1(unsigned int* buf, const float* __restrict__ w1e,
                                         int fc, int ks, int tid) {
    int f_l = tid;                                // 0..255 == FCHUNK
    const float* base = w1e + (long)ks * Ff + (fc + f_l);
    #pragma unroll 4
    for (int kp = 0; kp < 32; ++kp) {
        float a  = base[(long)(2 * kp) * Ff];
        float bq = base[(long)(2 * kp + 1) * Ff];
        buf[f_l * SW1_STRIDE + kp] = pack2bf(a, bq);
    }
}

__global__ void __launch_bounds__(256)
moe_ffn_fused(const float* __restrict__ x,  const float* __restrict__ W1,
              const float* __restrict__ b1, const float* __restrict__ W2,
              const float* __restrict__ b2, float* __restrict__ out) {
    extern __shared__ unsigned char smem_raw[];
    unsigned int* sX   = (unsigned int*)(smem_raw);                        // [16][516] uints
    __bf16*       sH   = (__bf16*)(smem_raw + SX_BYTES);                   // [16][264] bf16
    unsigned int* sHu  = (unsigned int*)(smem_raw + SX_BYTES);
    unsigned int* sW   = (unsigned int*)(smem_raw + SX_BYTES + SH_BYTES);  // W2 stage / union
    unsigned int* sW1a = sW;                                               // W1 ping
    unsigned int* sW1b = sW + (SW1_BYTES / 4);                             // W1 pong

    const int tid  = threadIdx.x;
    const int lane = tid & 31;
    const int wave = tid >> 5;            // 0..7

    const int bid  = blockIdx.x;          // E * (MPE/16) = 2048 blocks
    const int e    = bid >> 8;
    const int mblk = bid & 255;
    const int m0   = mblk * 16;
    const int bb   = m0 / CAP;            // tile never crosses batch (16 | 1024)
    const int c0   = m0 % CAP;
    const long rowbase = (long)bb * (NE * CAP) + (long)e * CAP + c0;

    const float* xrow = x  + rowbase * Dm;
    const float* w1e  = W1 + (long)e * Dm * Ff;
    const float* w2e  = W2 + (long)e * Ff * Dm;

    // ---- Stage X tile [16 x 1024] fp32 -> bf16 pairs in LDS (resident) ----
    #pragma unroll 4
    for (int q = 0; q < 16; ++q) {
        int idx = tid + 256 * q;                  // float4 index 0..4095
        int r   = idx >> 8;
        int c4  = idx & 255;
        float4 v = ((const float4*)(xrow + (long)r * Dm))[c4];
        sX[r * SX_STRIDE + c4 * 2]     = pack2bf(v.x, v.y);
        sX[r * SX_STRIDE + c4 * 2 + 1] = pack2bf(v.z, v.w);
    }
    __syncthreads();

    v8f yacc[8];
    #pragma unroll
    for (int i = 0; i < 8; ++i) yacc[i] = v8f{};

    const int arow = lane & 15;

    for (int fc = 0; fc < Ff; fc += FCHUNK) {     // 16 F-chunks
        // ========== Phase 1: H[16 x 256] = X @ W1[:, fc:fc+256], ping-pong stages ========
        v8f hacc[2] = {v8f{}, v8f{}};
        stage_w1(sW1a, w1e, fc, 0, tid);          // preload stage 0
        __syncthreads();
        for (int si = 0; si < Dm / 64; ++si) {    // 16 K-stages of 64
            const int ks = si * 64;
            unsigned int* cur = (si & 1) ? sW1b : sW1a;
            unsigned int* nxt = (si & 1) ? sW1a : sW1b;
            if (si + 1 < Dm / 64)                 // issue next stage's loads before compute
                stage_w1(nxt, w1e, fc, ks + 64, tid);
            #pragma unroll
            for (int k0 = 0; k0 < 64; k0 += 32) {
                v16bf af = load_frag_a(sX, SX_STRIDE, arow, (ks + k0) >> 1, lane);
                #pragma unroll
                for (int j = 0; j < 2; ++j) {     // A reused across 2 f-subtiles
                    v16bf bf = load_frag_b(cur, SW1_STRIDE,
                                           16 * (2 * wave + j) + (lane & 15), k0 >> 1, lane);
                    hacc[j] = __builtin_amdgcn_wmma_f32_16x16x32_bf16(
                                  false, af, false, bf, (short)0, hacc[j], false, false);
                }
            }
            __syncthreads();                      // single barrier per stage (2-deep pipeline)
        }
        // bias + gelu -> bf16 -> sH (wave owns 32 F-columns, all 16 rows)
        {
            int rhi = ((lane >> 4) & 1) * 8;
            #pragma unroll
            for (int j = 0; j < 2; ++j) {
                int   col  = 32 * wave + 16 * j + (lane & 15);
                float bias = b1[(long)e * Ff + fc + col];
                #pragma unroll
                for (int v = 0; v < 8; ++v) {
                    float h = gelu_tanh(hacc[j][v] + bias);
                    sH[(v + rhi) * SH_STRIDE_H + col] = (__bf16)h;
                }
            }
        }
        __syncthreads();

        // ========== Phase 2: Y[16 x 1024] += H[16 x 256] @ W2[fc:fc+256, :] =============
        for (int k2s = 0; k2s < FCHUNK; k2s += 32) {  // 8 K-stages
            // stage W2 [32k x 1024n] transposed -> sW[n][kpair]  (coalesced over n)
            #pragma unroll 4
            for (int q = 0; q < 64; ++q) {
                int idx = tid + 256 * q;          // 0..16383
                int kp  = idx >> 10;
                int n   = idx & 1023;
                const float* base = w2e + (long)(fc + k2s + 2 * kp) * Dm + n;
                sW[n * SW2_STRIDE + kp] = pack2bf(base[0], base[Dm]);
            }
            // prefetch next W2 stage (or next chunk's first stage) into L2/WGP$
            {
                long nk = (k2s + 32 < FCHUNK) ? (long)(fc + k2s + 32) : (long)(fc + FCHUNK);
                if (nk < Ff) {
                    #pragma unroll
                    for (int p = 0; p < 4; ++p) {
                        int id  = tid + 256 * p;  // 1024 x 128B lines = 32 rows x 4KB
                        int row = id >> 5;
                        int cl  = id & 31;
                        __builtin_prefetch(w2e + (nk + row) * Dm + cl * 32, 0, 1);
                    }
                }
            }
            __syncthreads();
            v16bf af = load_frag_a(sHu, SH_STRIDE_U, arow, k2s >> 1, lane);
            #pragma unroll
            for (int nt = 0; nt < 8; ++nt) {      // A reused across 8 n-subtiles
                int nrow = 128 * wave + 16 * nt + (lane & 15);
                v16bf bf = load_frag_b(sW, SW2_STRIDE, nrow, 0, lane);
                yacc[nt] = __builtin_amdgcn_wmma_f32_16x16x32_bf16(
                               false, af, false, bf, (short)0, yacc[nt], false, false);
            }
            __syncthreads();
        }
    }

    // ---- Epilogue: + b2, store (C/D layout: lane = N, vgpr v = row v (+8 hi lanes)) ----
    {
        int rhi = ((lane >> 4) & 1) * 8;
        #pragma unroll
        for (int nt = 0; nt < 8; ++nt) {
            int   col   = 128 * wave + 16 * nt + (lane & 15);
            float bias2 = b2[(long)e * Dm + col];
            #pragma unroll
            for (int v = 0; v < 8; ++v) {
                out[(rowbase + v + rhi) * Dm + col] = yacc[nt][v] + bias2;
            }
        }
    }
}

extern "C" void kernel_launch(void* const* d_in, const int* in_sizes, int n_in,
                              void* d_out, int out_size, void* d_ws, size_t ws_size,
                              hipStream_t stream) {
    (void)in_sizes; (void)n_in; (void)d_ws; (void)ws_size; (void)out_size;
    const float* x  = (const float*)d_in[0];
    const float* W1 = (const float*)d_in[1];
    const float* b1 = (const float*)d_in[2];
    const float* W2 = (const float*)d_in[3];
    const float* b2 = (const float*)d_in[4];
    float* out = (float*)d_out;

    hipFuncSetAttribute((const void*)moe_ffn_fused,
                        hipFuncAttributeMaxDynamicSharedMemorySize, SMEM_BYTES);

    dim3 grid(NE * (MPE / 16));   // 2048 workgroups
    dim3 block(256);              // 8 wave32 waves
    moe_ffn_fused<<<grid, block, SMEM_BYTES, stream>>>(x, W1, b1, W2, b2, out);
}
